// PredCorrNetwork_57389353009415
// MI455X (gfx1250) — compile-verified, matrix operand order checked by reading, not run on hardware
//
#include <hip/hip_runtime.h>
#include <stdint.h>

// ---------------------------------------------------------------------------
// PredCorrNetwork rollout on gfx1250 (MI455X): all MLP GEMMs via
// v_wmma_f32_16x16x32_bf16. Activations in per-wave LDS; weights pre-packed
// bf16 column-major in d_ws and staged per 64-column chunk into LDS
// (async-to-LDS when available). ~0.66 TFLOP total, matrix-compute bound.
// ---------------------------------------------------------------------------

typedef __attribute__((ext_vector_type(16))) __bf16 v16bf;
typedef __attribute__((ext_vector_type(8)))  float  v8f;
typedef __attribute__((ext_vector_type(4)))  int    v4i;

#define AS1 __attribute__((address_space(1)))
#define AS3 __attribute__((address_space(3)))

#if defined(__has_builtin)
#if __has_builtin(__builtin_amdgcn_global_load_async_to_lds_b128)
#define USE_ASYNC_STAGE 1
#endif
#endif

#define NPAR      131072
#define NSTEPS    20
#define NSUB      5
#define DT        0.01f
#define GAMMA_    0.99f
#define UMAX      2.0f
#define C1_       0.8f
#define C2_       0.001f
#define C3_       0.8f

#define MBLK      256              // particles per block (1 per thread)
#define NWAVE     8                // 256 threads = 8 wave32
#define NBLOCKS   (NPAR / MBLK)    // 512

// ---- packed bf16 weight offsets inside d_ws (bf16 element units) ----------
// Packed layout per layer (column-major pairs): dword (n, kp) at index
// n*(KL/2)+kp holds {W[n][2kp], W[n][2kp+1]}  (zero-padded K).
#define OFF_S0    0
#define SZ_S0     (32*256)         // 4->256, K padded 4->32
#define OFF_S1    (OFF_S0 + SZ_S0)
#define SZ_S1     (256*256)
#define OFF_S2    (OFF_S1 + SZ_S1)
#define SZ_S2     (256*128)
#define OFF_A1    (OFF_S2 + SZ_S2)
#define SZ_A1     (128*128)
#define OFF_C0    (OFF_A1 + SZ_A1)
#define SZ_C0     (128*64)
#define OFF_C1    (OFF_C0 + SZ_C0)
#define SZ_C1     (64*32)
#define WS_WEIGHT_ELEMS (OFF_C1 + SZ_C1)   // 133120 bf16 = 266240 bytes

// ---- d_out layout (floats): x, y, loss, fid_store, act_store --------------
#define OUT_X     0
#define OUT_Y     (2*NPAR)
#define OUT_LOSS  (4*NPAR)
#define OUT_FID   (4*NPAR + 1)
#define OUT_ACT   (OUT_FID + NSTEPS*NPAR)

// ---- dynamic LDS layout ----------------------------------------------------
#define LDS_BUF_ELEMS  (NWAVE*32*256)      // 65536 bf16 = 128KB per buffer
#define LDS_STAGE_DW   8192                // 32KB: covers (256/2)*64 dwords
#define SMEM_BYTES     (2*LDS_BUF_ELEMS*2 + LDS_STAGE_DW*4)   // 294912 B

// ---------------------------------------------------------------------------
// bf16 pair helpers (dword = two bf16)
// ---------------------------------------------------------------------------
__device__ inline uint32_t pack_bf16(float lo, float hi) {
  union { __bf16 h; uint16_t s; } a, b;
  a.h = (__bf16)lo; b.h = (__bf16)hi;
  return ((uint32_t)b.s << 16) | (uint32_t)a.s;
}
__device__ inline float bf16_lo(uint32_t u) {
  union { uint32_t u; float f; } t; t.u = u << 16; return t.f;
}
__device__ inline float bf16_hi(uint32_t u) {
  union { uint32_t u; float f; } t; t.u = u & 0xffff0000u; return t.f;
}

// ---------------------------------------------------------------------------
// Fragment helpers (layouts per CDNA5 ISA 7.12.2, 16-bit A 16x32 / B 32x16).
// Both fragments are two contiguous 16B runs -> ds_load_b128 pairs.
// ---------------------------------------------------------------------------
union FragBF { v16bf v; uint32_t u[8]; uint4 q[2]; };

__device__ inline v8f zero8() {
  v8f z;
#pragma unroll
  for (int i = 0; i < 8; ++i) z[i] = 0.0f;
  return z;
}

// A fragment: 16x32 bf16 activation tile, rows = this wave's particles.
__device__ inline v16bf load_fragA(const __bf16* inBase, int lane, int mt, int kc) {
  const int m  = (lane & 15) + mt * 16;
  const int hi = (lane >> 4) & 1;
  const __bf16* p = inBase + m * 256 + kc * 32 + (hi ? 8 : 0);
  FragBF f;
  f.q[0] = *(const uint4*)(p);        // K = kc*32 + hiK + 0..7
  f.q[1] = *(const uint4*)(p + 16);   // K = kc*32 + hiK + 16..23
  return f.v;
}

// B fragment: 32x16 weight tile from LDS stage (column-major pair layout).
template<int KL>
__device__ inline v16bf load_fragB(const uint32_t* stageDW, int lane, int nt, int kc) {
  const int n = (lane & 15) + nt * 16;          // column within staged chunk
  const int h = ((lane >> 4) & 1) ? 4 : 0;      // hiK/2
  const uint32_t* p = stageDW + n * (KL / 2) + kc * 16 + h;
  FragBF f;
  f.q[0] = *(const uint4*)(p);        // k pairs: kc*32+hiK+{0,2,4,6}
  f.q[1] = *(const uint4*)(p + 8);    // k pairs: kc*32+16+hiK+{0,2,4,6}
  return f.v;
}

// ---------------------------------------------------------------------------
// Generic WMMA dense layer: out[0:NL) = relu(in[0:KL) @ W.T + b)
// Weights pre-packed bf16 column-major in d_ws; staged per NC-column chunk
// into LDS (contiguous copy; async-to-LDS when the builtin exists).
// Per wave: 2 M-tiles (its 32 particles). Block-uniform control flow.
// ---------------------------------------------------------------------------
template<int KL, int NL, int NC>
__device__ void mlp_layer(const __bf16* __restrict__ wpk,  // packed weights (global)
                          const float*  __restrict__ bias, // f32 bias (global)
                          const __bf16* inBase,            // wave act in, stride 256
                          __bf16*       outBase,           // wave act out, stride 256
                          int outOff,
                          uint32_t* stage, int lane, int tid)
{
  constexpr int KC = KL / 32;            // k-chunks of 32
  constexpr int NT = NC / 16;            // n-tiles per chunk (<=4)
  constexpr int CHUNK_DW = (KL / 2) * NC;
  constexpr int TOT4 = CHUNK_DW / 4;     // 16B packets per chunk (multiple of 256)
  const uint32_t* Wdw = (const uint32_t*)wpk;

  for (int nc = 0; nc < NL / NC; ++nc) {
    __syncthreads();                     // stage buffer free to overwrite
    {
      const uint4* src4 = (const uint4*)(Wdw + nc * CHUNK_DW);
      uint4*       dst4 = (uint4*)stage;
#ifdef USE_ASYNC_STAGE
#pragma unroll 1
      for (int d4 = tid; d4 < TOT4; d4 += MBLK) {
        __builtin_amdgcn_global_load_async_to_lds_b128(
            (AS1 v4i*)(src4 + d4), (AS3 v4i*)(dst4 + d4), 0, 0);
      }
#if __has_builtin(__builtin_amdgcn_s_wait_asynccnt)
      __builtin_amdgcn_s_wait_asynccnt(0);
#else
      asm volatile("s_wait_asynccnt 0" ::: "memory");
#endif
#else
#pragma unroll 1
      for (int d4 = tid; d4 < TOT4; d4 += MBLK) dst4[d4] = src4[d4];
#endif
    }
    __syncthreads();                     // stage ready for all waves

    v8f acc[2][NT];
#pragma unroll
    for (int mt = 0; mt < 2; ++mt)
#pragma unroll
      for (int nt = 0; nt < NT; ++nt) acc[mt][nt] = zero8();

    for (int kc = 0; kc < KC; ++kc) {
      const v16bf a0 = load_fragA(inBase, lane, 0, kc);
      const v16bf a1 = load_fragA(inBase, lane, 1, kc);
#pragma unroll
      for (int nt = 0; nt < NT; ++nt) {
        const v16bf b = load_fragB<KL>(stage, lane, nt, kc);
        acc[0][nt] = __builtin_amdgcn_wmma_f32_16x16x32_bf16(
            false, a0, false, b, (short)0, acc[0][nt], false, false);
        acc[1][nt] = __builtin_amdgcn_wmma_f32_16x16x32_bf16(
            false, a1, false, b, (short)0, acc[1][nt], false, false);
      }
    }

    // epilogue: bias + relu, bf16 store back to wave's activation buffer.
    // D layout: lane holds column n = lane&15; rows m = (lane>>4)*8 + r.
    const int nl = lane & 15;
    const int mb = (lane >> 4) * 8;
#pragma unroll
    for (int mt = 0; mt < 2; ++mt) {
#pragma unroll
      for (int nt = 0; nt < NT; ++nt) {
        const int ng = nc * NC + nt * 16 + nl;
        const float bv = bias[ng];
#pragma unroll
        for (int r = 0; r < 8; ++r) {
          float vv = acc[mt][nt][r] + bv;
          vv = fmaxf(vv, 0.0f);
          outBase[(mt * 16 + mb + r) * 256 + outOff + ng] = (__bf16)vv;
        }
      }
    }
  }
}

// ---------------------------------------------------------------------------
// Pre-pass: pack f32 row-major W[N][K] -> bf16 column-major K-pairs, K padded.
// dword d = n*(KLpad/2)+kp : {W[n][2kp], W[n][2kp+1]} (0 beyond KLsrc).
// ---------------------------------------------------------------------------
__global__ void pack_w(const float* __restrict__ W, uint32_t* __restrict__ dst,
                       int KLsrc, int KLpad, int NL) {
  const int d = blockIdx.x * blockDim.x + threadIdx.x;
  const int kph = KLpad >> 1;
  const int total = kph * NL;
  if (d >= total) return;
  const int n  = d / kph;
  const int kp = d - n * kph;
  const int k0 = 2 * kp, k1 = k0 + 1;
  const float lo = (k0 < KLsrc) ? W[n * KLsrc + k0] : 0.0f;
  const float hi = (k1 < KLsrc) ? W[n * KLsrc + k1] : 0.0f;
  dst[d] = pack_bf16(lo, hi);
}

// ---------------------------------------------------------------------------
// Main rollout kernel
// ---------------------------------------------------------------------------
__global__ __launch_bounds__(MBLK)
void rollout_kernel(const float* __restrict__ psi_x, const float* __restrict__ psi_y,
                    const __bf16* __restrict__ wpk,
                    const float* __restrict__ bS0, const float* __restrict__ bS1,
                    const float* __restrict__ bS2,
                    const float* __restrict__ wA0, const float* __restrict__ bA0,
                    const float* __restrict__ bA1,
                    const float* __restrict__ bC0, const float* __restrict__ bC1,
                    const float* __restrict__ wC2, const float* __restrict__ bC2,
                    float* __restrict__ out, float* __restrict__ partials)
{
  extern __shared__ __align__(16) char smem[];
  __bf16*   bufA  = (__bf16*)smem;                       // [8][32][256]
  __bf16*   bufB  = bufA + LDS_BUF_ELEMS;                // [8][32][256]
  uint32_t* stage = (uint32_t*)(bufB + LDS_BUF_ELEMS);   // 8192 dwords

  const int tid  = threadIdx.x;
  const int wave = tid >> 5;
  const int lane = tid & 31;
  __bf16* myA = bufA + wave * (32 * 256);
  __bf16* myB = bufB + wave * (32 * 256);
  uint32_t* rowA = (uint32_t*)(myA + lane * 256);        // this particle's row (dwords)
  uint32_t* rowB = (uint32_t*)(myB + lane * 256);
  const int p = blockIdx.x * MBLK + wave * 32 + lane;

  float x0 = psi_x[2 * p + 0], x1 = psi_x[2 * p + 1];
  float y0 = psi_y[2 * p + 0], y1 = psi_y[2 * p + 1];
  float alpha = 0.0f, gj = 1.0f, lossAcc = 0.0f, fid = 0.0f;

  for (int j = 0; j < NSTEPS; ++j) {
    // ---- state-net input [x, y], zero-padded to K=32 (dword stores) ----
    rowA[0] = pack_bf16(x0, x1);
    rowA[1] = pack_bf16(y0, y1);
#pragma unroll
    for (int i = 2; i < 16; ++i) rowA[i] = 0u;

    mlp_layer<32, 256, 64>(wpk + OFF_S0, bS0, myA, myB, 0,   stage, lane, tid); // 4->256
    mlp_layer<256,256, 64>(wpk + OFF_S1, bS1, myB, myA, 0,   stage, lane, tid); // 256->256
    mlp_layer<256,128, 64>(wpk + OFF_S2, bS2, myA, myB, 0,   stage, lane, tid); // d1 in bufB[0:128)

    // ---- action-net layer 0 (scalar input): per-thread, paired stores ----
    const float an = alpha * (1.0f / UMAX);
#pragma unroll 4
    for (int n = 0; n < 128; n += 2) {
      const float v0 = fmaxf(wA0[n + 0] * an + bA0[n + 0], 0.0f);
      const float v1 = fmaxf(wA0[n + 1] * an + bA0[n + 1], 0.0f);
      rowA[n >> 1] = pack_bf16(v0, v1);                  // a0 in bufA[0:128)
    }
    mlp_layer<128,128, 64>(wpk + OFF_A1, bA1, myA, myA, 128, stage, lane, tid); // d2 in bufA[128:256)

    // ---- combine input: d1 + d2 (dword-paired) ----
#pragma unroll 4
    for (int i = 0; i < 64; ++i) {
      const uint32_t ub = rowB[i];
      const uint32_t ua = rowA[64 + i];
      rowA[i] = pack_bf16(bf16_lo(ub) + bf16_lo(ua),
                          bf16_hi(ub) + bf16_hi(ua));    // dsum in bufA[0:128)
    }
    mlp_layer<128, 64, 64>(wpk + OFF_C0, bC0, myA, myB, 0,   stage, lane, tid); // c0 in bufB[0:64)
    mlp_layer< 64, 32, 32>(wpk + OFF_C1, bC1, myB, myA, 0,   stage, lane, tid); // c1 in bufA[0:32)

    // ---- combine final 32->1 (per-thread), soft-sign, clip ----
    float d = bC2[0];
#pragma unroll
    for (int i = 0; i < 16; ++i) {
      const uint32_t u = rowA[i];
      d += bf16_lo(u) * wC2[2 * i + 0];
      d += bf16_hi(u) * wC2[2 * i + 1];
    }
    alpha = UMAX * (d / (1.0f + fabsf(d)));
    alpha = fminf(fmaxf(alpha, -UMAX), UMAX);

    // ---- 5 Heun substeps of the 2x2 Hamiltonian on (x, y) ----
    const float h00 = 0.5f * DT;         // W_FREQ/2 * DT
    const float h01 = alpha * DT;        // off-diagonal * DT
#pragma unroll
    for (int s = 0; s < NSUB; ++s) {
      const float fx0 =  (h00 * y0 + h01 * y1);
      const float fx1 =  (h01 * y0 - h00 * y1);
      const float fy0 = -(h00 * x0 + h01 * x1);
      const float fy1 = -(h01 * x0 - h00 * x1);
      const float xt0 = x0 + fx0, xt1 = x1 + fx1;
      const float yt0 = y0 + fy0, yt1 = y1 + fy1;
      const float nx0 = x0 + 0.5f * ( (h00 * yt0 + h01 * yt1) + fx0);
      const float nx1 = x1 + 0.5f * ( (h01 * yt0 - h00 * yt1) + fx1);
      const float ny0 = y0 + 0.5f * (-(h00 * xt0 + h01 * xt1) + fy0);
      const float ny1 = y1 + 0.5f * (-(h01 * xt0 - h00 * xt1) + fy1);
      x0 = nx0; x1 = nx1; y0 = ny0; y1 = ny1;
    }

    fid = x0 * x0 + y0 * y0;             // target = [1, 0]
    out[OUT_FID + j * NPAR + p] = fid;
    out[OUT_ACT + j * NPAR + p] = alpha;
    lossAcc += C1_ * gj * (1.0f - fid) + C2_ * alpha * alpha;
    gj *= GAMMA_;
  }

  out[OUT_X + 2 * p + 0] = x0;  out[OUT_X + 2 * p + 1] = x1;
  out[OUT_Y + 2 * p + 0] = y0;  out[OUT_Y + 2 * p + 1] = y1;
  lossAcc += C3_ * (1.0f - fid);

  // ---- deterministic block tree-reduction of loss (reuse stage as f32) ----
  float* red = (float*)stage;
  __syncthreads();
  red[tid] = lossAcc;
  __syncthreads();
  for (int off = MBLK / 2; off > 0; off >>= 1) {
    if (tid < off) red[tid] += red[tid + off];
    __syncthreads();
  }
  if (tid == 0) partials[blockIdx.x] = red[0];
}

// ---------------------------------------------------------------------------
// Final loss reduction: mean over all particles
// ---------------------------------------------------------------------------
__global__ void reduce_loss(const float* __restrict__ partials, int n,
                            float* __restrict__ outLoss) {
  __shared__ float s[256];
  float v = 0.0f;
  for (int i = threadIdx.x; i < n; i += 256) v += partials[i];
  s[threadIdx.x] = v;
  __syncthreads();
  for (int off = 128; off > 0; off >>= 1) {
    if (threadIdx.x < off) s[threadIdx.x] += s[threadIdx.x + off];
    __syncthreads();
  }
  if (threadIdx.x == 0) outLoss[0] = s[0] * (1.0f / (float)NPAR);
}

// ---------------------------------------------------------------------------
// Launch. Input order (setup_inputs dict order; params flattened with dict
// keys sorted: action, combine, state; each layer as (W, b)):
//  0 psi_x  1 psi_y
//  2 A0.W(128x1)   3 A0.b(128)   4 A1.W(128x128) 5 A1.b(128)
//  6 C0.W(64x128)  7 C0.b(64)    8 C1.W(32x64)   9 C1.b(32)
// 10 C2.W(1x32)   11 C2.b(1)
// 12 S0.W(256x4)  13 S0.b(256)  14 S1.W(256x256) 15 S1.b(256)
// 16 S2.W(128x256)17 S2.b(128)
// ---------------------------------------------------------------------------
extern "C" void kernel_launch(void* const* d_in, const int* in_sizes, int n_in,
                              void* d_out, int out_size, void* d_ws, size_t ws_size,
                              hipStream_t stream) {
  (void)in_sizes; (void)n_in; (void)out_size; (void)ws_size;

  const float* psi_x = (const float*)d_in[0];
  const float* psi_y = (const float*)d_in[1];
  const float* wA0 = (const float*)d_in[2];
  const float* bA0 = (const float*)d_in[3];
  const float* wA1 = (const float*)d_in[4];
  const float* bA1 = (const float*)d_in[5];
  const float* wC0 = (const float*)d_in[6];
  const float* bC0 = (const float*)d_in[7];
  const float* wC1 = (const float*)d_in[8];
  const float* bC1 = (const float*)d_in[9];
  const float* wC2 = (const float*)d_in[10];
  const float* bC2 = (const float*)d_in[11];
  const float* wS0 = (const float*)d_in[12];
  const float* bS0 = (const float*)d_in[13];
  const float* wS1 = (const float*)d_in[14];
  const float* bS1 = (const float*)d_in[15];
  const float* wS2 = (const float*)d_in[16];
  const float* bS2 = (const float*)d_in[17];

  __bf16* wpk     = (__bf16*)d_ws;
  float* partials = (float*)((char*)d_ws + (size_t)WS_WEIGHT_ELEMS * 2);
  float* out      = (float*)d_out;

  // pack all GEMM weights to bf16 column-major pairs (zero-padded K) in d_ws
  struct PackJob { const float* W; int off, ks, kp, nl; };
  const PackJob jobs[6] = {
    { wS0, OFF_S0,   4,  32, 256 },
    { wS1, OFF_S1, 256, 256, 256 },
    { wS2, OFF_S2, 256, 256, 128 },
    { wA1, OFF_A1, 128, 128, 128 },
    { wC0, OFF_C0, 128, 128,  64 },
    { wC1, OFF_C1,  64,  64,  32 },
  };
  for (int i = 0; i < 6; ++i) {
    const int total  = (jobs[i].kp / 2) * jobs[i].nl;
    const int blocks = (total + 255) / 256;
    pack_w<<<blocks, 256, 0, stream>>>(jobs[i].W, (uint32_t*)(wpk + jobs[i].off),
                                       jobs[i].ks, jobs[i].kp, jobs[i].nl);
  }

  (void)hipFuncSetAttribute(reinterpret_cast<const void*>(&rollout_kernel),
                            hipFuncAttributeMaxDynamicSharedMemorySize, SMEM_BYTES);

  rollout_kernel<<<NBLOCKS, MBLK, SMEM_BYTES, stream>>>(
      psi_x, psi_y, wpk, bS0, bS1, bS2, wA0, bA0, bA1, bC0, bC1, wC2, bC2,
      out, partials);

  reduce_loss<<<1, 256, 0, stream>>>(partials, NBLOCKS, out + OUT_LOSS);
}
